// SeaReader_4260607557905
// MI455X (gfx1250) — compile-verified, matrix operand order checked by reading, not run on hardware
//
#include <hip/hip_runtime.h>
#include <hip/hip_bf16.h>

// ---------------------------------------------------------------------------
// SeaReader (B=4, O=4, L=128, N=8, V=256, H=256) for MI455X / gfx1250.
// All matmuls use v_wmma_f32_16x16x32_bf16. All WMMA operands are bf16 in
// memory (converted once); fragment loads are 2x b128 per lane. GEMMs use a
// 2x2 register macro-tile (4 WMMA per 4 fragment loads). GRU scans are fused
// persistent-tile kernels with h resident in LDS and K-outer fragment reuse.
// ---------------------------------------------------------------------------

typedef __attribute__((ext_vector_type(8)))  float  v8f;
typedef __attribute__((ext_vector_type(16))) __bf16 v16bf;

__device__ __forceinline__ unsigned short f2bf(float f) {
  union { __bf16 b; unsigned short s; } u;
  u.b = (__bf16)f;                       // hardware RNE convert
  return u.s;
}

// 16 bf16 (32 bytes) per lane -> two b128 loads (works for global and LDS ptrs)
__device__ __forceinline__ v16bf load_frag(const unsigned short* p) {
  union { uint4 q[2]; v16bf v; } u;
  const uint4* q = (const uint4*)p;
  u.q[0] = q[0];
  u.q[1] = q[1];
  return u.v;
}

__device__ __forceinline__ float sigmoidf_(float x) {
  return 1.f / (1.f + __expf(-x));
}

// ---------------------------------------------------------------------------
// f32 -> bf16 streaming convert (4 elems/thread).
// ---------------------------------------------------------------------------
__global__ __launch_bounds__(256) void cvt_bf16_k(
    const float* __restrict__ in, unsigned short* __restrict__ outp, long long n4)
{
  const long long i = (long long)blockIdx.x * 256 + threadIdx.x;
  if (i >= n4) return;
  const float4 f = ((const float4*)in)[i];
  union { unsigned short s[4]; uint2 u; } r;
  r.s[0] = f2bf(f.x); r.s[1] = f2bf(f.y); r.s[2] = f2bf(f.z); r.s[3] = f2bf(f.w);
  ((uint2*)outp)[i] = r.u;
}

// ---------------------------------------------------------------------------
// Generic batched GEMM: C[batch] = A[batch/aDiv] (MxK) x Bt[batch/bDiv]^T,
// A and Bt bf16 row-major with K contiguous. 2x2 macro-tile (32x32 C) per
// wave, 8 waves/block. M,N multiples of 32; K of 32. C f32 or bf16.
// ---------------------------------------------------------------------------
__global__ __launch_bounds__(256) void gemm_bf16_wmma(
    const unsigned short* __restrict__ A, const unsigned short* __restrict__ Bt,
    void* __restrict__ C, int N, int K,
    long long sA, long long sB, long long sC,
    int aDiv, int bDiv, int tiles32PerBatch, int cBf16)
{
  const int wave = threadIdx.x >> 5, lane = threadIdx.x & 31;
  const int tile = blockIdx.x * 8 + wave;
  if (tile >= tiles32PerBatch) return;        // uniform per wave
  const int ntN = N >> 5;
  const int tm = (tile / ntN) << 5;           // 32-row macro tile base
  const int tn = (tile % ntN) << 5;           // 32-col macro tile base
  const int batch = blockIdx.y;
  const int lr = lane & 15;
  const int koff = (lane >> 4) << 4;

  const unsigned short* Ar0 = A + (long long)(batch / aDiv) * sA
                                + (long long)(tm + lr) * K + koff;
  const unsigned short* Ar1 = Ar0 + (long long)16 * K;
  const unsigned short* Br0 = Bt + (long long)(batch / bDiv) * sB
                                + (long long)(tn + lr) * K + koff;
  const unsigned short* Br1 = Br0 + (long long)16 * K;

  const v8f vz = {0.f, 0.f, 0.f, 0.f, 0.f, 0.f, 0.f, 0.f};
  v8f c00 = vz, c01 = vz, c10 = vz, c11 = vz;
  for (int k = 0; k < K; k += 32) {
    v16bf a0 = load_frag(Ar0 + k);
    v16bf a1 = load_frag(Ar1 + k);
    v16bf b0 = load_frag(Br0 + k);
    v16bf b1 = load_frag(Br1 + k);
    c00 = __builtin_amdgcn_wmma_f32_16x16x32_bf16(false, a0, false, b0, (short)0, c00, false, false);
    c01 = __builtin_amdgcn_wmma_f32_16x16x32_bf16(false, a0, false, b1, (short)0, c01, false, false);
    c10 = __builtin_amdgcn_wmma_f32_16x16x32_bf16(false, a1, false, b0, (short)0, c10, false, false);
    c11 = __builtin_amdgcn_wmma_f32_16x16x32_bf16(false, a1, false, b1, (short)0, c11, false, false);
  }

  const int rb = (lane >> 4) << 3;            // C: VGPR i -> row i + 8*(lane/16)
  if (cBf16) {
    unsigned short* Cb = (unsigned short*)C + (long long)batch * sC + lr;
#pragma unroll
    for (int i = 0; i < 8; ++i) {
      Cb[(long long)(tm + rb + i) * N + tn]           = f2bf(c00[i]);
      Cb[(long long)(tm + rb + i) * N + tn + 16]      = f2bf(c01[i]);
      Cb[(long long)(tm + 16 + rb + i) * N + tn]      = f2bf(c10[i]);
      Cb[(long long)(tm + 16 + rb + i) * N + tn + 16] = f2bf(c11[i]);
    }
  } else {
    float* Cf = (float*)C + (long long)batch * sC + lr;
#pragma unroll
    for (int i = 0; i < 8; ++i) {
      Cf[(long long)(tm + rb + i) * N + tn]           = c00[i];
      Cf[(long long)(tm + rb + i) * N + tn + 16]      = c01[i];
      Cf[(long long)(tm + 16 + rb + i) * N + tn]      = c10[i];
      Cf[(long long)(tm + 16 + rb + i) * N + tn + 16] = c11[i];
    }
  }
}

// ---------------------------------------------------------------------------
// Fused GRU scan (gate order r,z,n; zero-init hidden; H=256).
// Block = 8 waves handles 16 sequences for all T steps. Wave w owns hidden
// columns [32w,32w+32) for all 3 gates (wave-local nonlinearity).
// K-outer loops: each A-fragment (x_t / h) loaded once, reused by 6 WMMAs.
// Weight-row pointers + biases precomputed once outside the time loop.
// ---------------------------------------------------------------------------
#define XPITCH 528   // 512 + 16 bf16 pad  (1056 B rows: 16B aligned, bank-rotated)
#define HPITCH 272   // 256 + 16 bf16 pad  (544 B rows)
__global__ __launch_bounds__(256) void gru_wmma(
    const unsigned short* __restrict__ x,    // [seqs, T, D] bf16
    const unsigned short* __restrict__ wih,  // [768, D] bf16
    const unsigned short* __restrict__ whh,  // [768, 256] bf16
    const float* __restrict__ bih,           // [768]
    const float* __restrict__ bhh,           // [768]
    unsigned short* __restrict__ outAll,     // [seqs, T, 256] bf16 or null
    float* __restrict__ outMax,              // [seqs, 256] or null (max over t)
    int T, int D)
{
  __shared__ float          hF[16][257];     // hidden f32 (recurrence precision)
  __shared__ unsigned short hB[16][HPITCH];  // hidden bf16 (A-frags)
  __shared__ unsigned short xB[16][XPITCH];  // x_t bf16 (A-frags)

  const int tid  = threadIdx.x;
  const int lane = tid & 31, wave = tid >> 5;
  const int lr   = lane & 15;
  const int koff = (lane >> 4) << 4;
  const int rb   = (lane >> 4) << 3;
  const int seq0 = blockIdx.x << 4;
  const long long seqStride = (long long)T * D;

  for (int i = tid; i < 16 * 256; i += 256) {
    hF[i >> 8][i & 255] = 0.f;
    hB[i >> 8][i & 255] = 0;
  }
  __syncthreads();

  // Per-(subtile,gate) weight-row base pointers and biases (loop-invariant).
  const unsigned short* wp[2][3];
  const unsigned short* hp[2][3];
  float bx[2][3], bh[2][3];
#pragma unroll
  for (int s = 0; s < 2; ++s)
#pragma unroll
    for (int g = 0; g < 3; ++g) {
      const int nrow = g * 256 + wave * 32 + s * 16 + lr;
      wp[s][g] = wih + (long long)nrow * D + koff;
      hp[s][g] = whh + (long long)nrow * 256 + koff;
      bx[s][g] = bih[nrow];
      bh[s][g] = bhh[nrow];
    }

  float hmax[2][8];
#pragma unroll
  for (int s = 0; s < 2; ++s)
#pragma unroll
    for (int i = 0; i < 8; ++i) hmax[s][i] = -3.4e38f;

  const int d8 = D >> 3;                    // uint4 chunks per row

  for (int t = 0; t < T; ++t) {
    // stage x_t tile [16, D] bf16 into LDS; prefetch next step's rows
    const unsigned short* xt = x + (long long)seq0 * seqStride + (long long)t * D;
    for (int i = tid; i < 16 * d8; i += 256) {
      const int m = i / d8, c8 = (i - m * d8) << 3;
      const unsigned short* src = &xt[(long long)m * seqStride + c8];
      *(uint4*)&xB[m][c8] = *(const uint4*)src;
      __builtin_prefetch(src + D, 0, 0);    // next t -> global_prefetch_b8
    }
    __syncthreads();

    v8f accX[2][3], accH[2][3];
#pragma unroll
    for (int s = 0; s < 2; ++s)
#pragma unroll
      for (int g = 0; g < 3; ++g) {
#pragma unroll
        for (int i = 0; i < 8; ++i) { accX[s][g][i] = bx[s][g]; accH[s][g][i] = bh[s][g]; }
      }

    // x_t @ Wih^T : A-fragment loaded once per k, reused by all 6 tiles
    for (int k = 0; k < D; k += 32) {
      const v16bf af = load_frag(&xB[lr][k + koff]);
#pragma unroll
      for (int s = 0; s < 2; ++s)
#pragma unroll
        for (int g = 0; g < 3; ++g) {
          const v16bf bfr = load_frag(wp[s][g] + k);
          accX[s][g] = __builtin_amdgcn_wmma_f32_16x16x32_bf16(
              false, af, false, bfr, (short)0, accX[s][g], false, false);
        }
    }
    // h @ Whh^T
    for (int k = 0; k < 256; k += 32) {
      const v16bf af = load_frag(&hB[lr][k + koff]);
#pragma unroll
      for (int s = 0; s < 2; ++s)
#pragma unroll
        for (int g = 0; g < 3; ++g) {
          const v16bf bfr = load_frag(hp[s][g] + k);
          accH[s][g] = __builtin_amdgcn_wmma_f32_16x16x32_bf16(
              false, af, false, bfr, (short)0, accH[s][g], false, false);
        }
    }
    __syncthreads();   // all waves done reading hB/xB before h update

#pragma unroll
    for (int s = 0; s < 2; ++s) {
      const int c = wave * 32 + s * 16 + lr;     // C layout: lane%16 = column
#pragma unroll
      for (int i = 0; i < 8; ++i) {              // VGPR i -> row i + 8*(lane/16)
        const int m = rb + i;
        const float r = sigmoidf_(accX[s][0][i] + accH[s][0][i]);
        const float z = sigmoidf_(accX[s][1][i] + accH[s][1][i]);
        const float n = tanhf(accX[s][2][i] + r * accH[s][2][i]);
        const float hn = (1.f - z) * n + z * hF[m][c];
        const unsigned short hb = f2bf(hn);
        hF[m][c] = hn;
        hB[m][c] = hb;
        if (outAll)
          outAll[((long long)(seq0 + m) * T + t) * 256 + c] = hb;
        hmax[s][i] = fmaxf(hmax[s][i], hn);
      }
    }
    __syncthreads();
  }

  if (outMax) {
#pragma unroll
    for (int s = 0; s < 2; ++s) {
      const int c = wave * 32 + s * 16 + lr;
#pragma unroll
      for (int i = 0; i < 8; ++i)
        outMax[(long long)(seq0 + rb + i) * 256 + c] = hmax[s][i];
    }
  }
}

// ---------------------------------------------------------------------------
// Batched 16x16-tiled transpose: in [batch,R,C] -> out [batch,C,R].
// ---------------------------------------------------------------------------
template <typename T>
__global__ __launch_bounds__(256) void transpose_k(
    const T* __restrict__ in, T* __restrict__ out, int R, int C)
{
  __shared__ T tbuf[16][17];
  const int tx = threadIdx.x & 15, ty = threadIdx.x >> 4;
  const long long base = (long long)blockIdx.z * R * C;
  const int r0 = blockIdx.y << 4, c0 = blockIdx.x << 4;
  tbuf[ty][tx] = in[base + (long long)(r0 + ty) * C + c0 + tx];
  __syncthreads();
  out[base + (long long)(c0 + ty) * R + r0 + tx] = tbuf[tx][ty];
}

// ---------------------------------------------------------------------------
// Register-resident row softmax: f32 in -> bf16 out. One wave per row.
// ---------------------------------------------------------------------------
template <int COLS>
__global__ __launch_bounds__(256) void softmax_bf16_k(
    const float* __restrict__ in, unsigned short* __restrict__ outp)
{
  constexpr int PER = COLS / 32;
  const long long row = (long long)blockIdx.x * 8 + (threadIdx.x >> 5);
  const int lane = threadIdx.x & 31;
  const float* p = in + row * COLS;
  float v[PER];
  float m = -3.4e38f;
#pragma unroll
  for (int i = 0; i < PER; ++i) { v[i] = p[lane + 32 * i]; m = fmaxf(m, v[i]); }
#pragma unroll
  for (int off = 16; off > 0; off >>= 1) m = fmaxf(m, __shfl_xor(m, off, 32));
  float s = 0.f;
#pragma unroll
  for (int i = 0; i < PER; ++i) { v[i] = __expf(v[i] - m); s += v[i]; }
#pragma unroll
  for (int off = 16; off > 0; off >>= 1) s += __shfl_xor(s, off, 32);
  const float inv = 1.f / s;
  unsigned short* q = outp + row * COLS;
#pragma unroll
  for (int i = 0; i < PER; ++i) q[lane + 32 * i] = f2bf(v[i] * inv);
}

// stmtcat_b[b,o,t,v] = bf16( t<128 ? statement[b,t,v] : answer[b,o,t-128,v] )
__global__ __launch_bounds__(256) void stmtcat_k(
    const float* __restrict__ stmt, const float* __restrict__ ans,
    unsigned short* __restrict__ outp)
{
  const int i = blockIdx.x * 256 + threadIdx.x;   // < 16*256*256
  const int v = i & 255;
  const int t = (i >> 8) & 255;
  const int bo = i >> 16;
  const int b = bo >> 2, o = bo & 3;
  const float val = (t < 128) ? stmt[(b * 128 + t) * 256 + v]
                              : ans[((b * 4 + o) * 128 + (t - 128)) * 256 + v];
  outp[i] = f2bf(val);
}

// dri[(bon,l), 0:256]=docs_enc, [256:512]=doc_read (all bf16)
__global__ __launch_bounds__(256) void concat_dri_k(
    const unsigned short* __restrict__ docs, const unsigned short* __restrict__ dread,
    unsigned short* __restrict__ dri)
{
  const long long i = (long long)blockIdx.x * 256 + threadIdx.x;  // < 128*128*512
  const int d = (int)(i & 511);
  const long long bl = i >> 9;
  dri[i] = (d < 256) ? docs[bl * 256 + d] : dread[bl * 256 + (d - 256)];
}

// Gate + pool + score: one block per (b,o).
__global__ __launch_bounds__(256) void finalize_k(
    const float* __restrict__ s_r, const float* __restrict__ d_r,
    const float* __restrict__ gate_w, const float* __restrict__ gate_b,
    const float* __restrict__ out_w, const float* __restrict__ out_b,
    float* __restrict__ logits)
{
  __shared__ float red[256];
  const int bo = blockIdx.x, t = threadIdx.x;
  float maxS = -3.4e38f, maxD = -3.4e38f, sumS = 0.f, sumD = 0.f;
  for (int n = 0; n < 8; ++n) {
    const float sv = s_r[((bo << 3) + n) * 256 + t];
    const float dv = d_r[((bo << 3) + n) * 256 + t];
    red[t] = sv * gate_w[t] + dv * gate_w[256 + t];
    __syncthreads();
    for (int off = 128; off > 0; off >>= 1) {
      if (t < off) red[t] += red[t + off];
      __syncthreads();
    }
    const float coef = red[0] + gate_b[0];
    __syncthreads();
    const float gs = coef * sv, gd = coef * dv;
    maxS = fmaxf(maxS, gs); maxD = fmaxf(maxD, gd);
    sumS += gs; sumD += gd;
  }
  red[t] = maxS * out_w[t] + maxD * out_w[256 + t]
         + (sumS * 0.125f) * out_w[512 + t] + (sumD * 0.125f) * out_w[768 + t];
  __syncthreads();
  for (int off = 128; off > 0; off >>= 1) {
    if (t < off) red[t] += red[t + off];
    __syncthreads();
  }
  if (t == 0) logits[bo] = red[0] + out_b[0];
}

__global__ void softmax_out_k(const float* __restrict__ logits, float* __restrict__ outp)
{
  const int b = threadIdx.x;
  if (b < 4) {
    float v[4]; float m = -3.4e38f;
    for (int o = 0; o < 4; ++o) { v[o] = logits[b * 4 + o]; m = fmaxf(m, v[o]); }
    float s = 0.f;
    for (int o = 0; o < 4; ++o) { v[o] = __expf(v[o] - m); s += v[o]; }
    for (int o = 0; o < 4; ++o) outp[b * 4 + o] = v[o] / s;
  }
}

// ---------------------------------------------------------------------------
extern "C" void kernel_launch(void* const* d_in, const int* in_sizes, int n_in,
                              void* d_out, int out_size, void* d_ws, size_t ws_size,
                              hipStream_t stream) {
  (void)in_sizes; (void)n_in; (void)out_size; (void)ws_size;
  const float* statement = (const float*)d_in[0];
  const float* answer    = (const float*)d_in[1];
  const float* refs      = (const float*)d_in[2];
  const float* ctx_wih = (const float*)d_in[3];
  const float* ctx_whh = (const float*)d_in[4];
  const float* ctx_bih = (const float*)d_in[5];
  const float* ctx_bhh = (const float*)d_in[6];
  const float* sr_wih  = (const float*)d_in[7];
  const float* sr_whh  = (const float*)d_in[8];
  const float* sr_bih  = (const float*)d_in[9];
  const float* sr_bhh  = (const float*)d_in[10];
  const float* dr_wih  = (const float*)d_in[11];
  const float* dr_whh  = (const float*)d_in[12];
  const float* dr_bih  = (const float*)d_in[13];
  const float* dr_bhh  = (const float*)d_in[14];
  const float* gate_w  = (const float*)d_in[15];
  const float* gate_b  = (const float*)d_in[16];
  const float* out_w   = (const float*)d_in[17];
  const float* out_b   = (const float*)d_in[18];

  char* ws = (char*)d_ws;
  size_t off = 0;
  auto allocF = [&](size_t elems) {
    float* p = (float*)(ws + off);
    off += ((elems * sizeof(float) + 255) & ~(size_t)255);
    return p;
  };
  auto allocB = [&](size_t elems) {
    unsigned short* p = (unsigned short*)(ws + off);
    off += ((elems * sizeof(unsigned short) + 255) & ~(size_t)255);
    return p;
  };

  // f32 intermediates (softmax inputs + small tails)
  float* match  = allocF(4194304);   // [128,256,128]
  float* matchT = allocF(4194304);   // [128,128,256]
  float* mm     = allocF(16777216);  // [16,1024,1024]
  float* s_r    = allocF(32768);     // [128,256]
  float* d_r    = allocF(32768);     // [128,256]
  float* logits = allocF(16);

  // bf16 operands
  unsigned short* stmtcat_b  = allocB(1048576);   // [16,256,256]
  unsigned short* refs_b     = allocB(4194304);   // [128,128,256]
  unsigned short* ctx_wih_b  = allocB(196608);    // [768,256]
  unsigned short* ctx_whh_b  = allocB(196608);
  unsigned short* sr_wih_b   = allocB(196608);
  unsigned short* sr_whh_b   = allocB(196608);
  unsigned short* dr_wih_b   = allocB(393216);    // [768,512]
  unsigned short* dr_whh_b   = allocB(196608);
  unsigned short* stmt_enc_b = allocB(1048576);   // [16,256,256]
  unsigned short* docs_enc_b = allocB(4194304);   // [128,128,256]
  unsigned short* stmt_encT_b= allocB(1048576);   // [16,256,256]
  unsigned short* docs_encT_b= allocB(4194304);   // [128,256,128]
  unsigned short* P1_b       = allocB(4194304);   // softmax(match) over l
  unsigned short* P2t_b      = allocB(4194304);   // softmax(match^T) over s
  unsigned short* read_sum_b = allocB(8388608);   // [128,256,256]
  unsigned short* doc_read_b = allocB(4194304);   // [128,128,256]
  unsigned short* dri_b      = allocB(8388608);   // [16,1024,512]
  unsigned short* driT_b     = allocB(8388608);   // [16,512,1024]
  unsigned short* P3_b       = allocB(16777216);  // softmax(mm) over m
  unsigned short* att_b      = allocB(8388608);   // [16,1024,512]

  // 0) one-time bf16 conversions (weights + inputs)
  cvt_bf16_k<<<192, 256, 0, stream>>>(ctx_wih, ctx_wih_b, 49152);
  cvt_bf16_k<<<192, 256, 0, stream>>>(ctx_whh, ctx_whh_b, 49152);
  cvt_bf16_k<<<192, 256, 0, stream>>>(sr_wih,  sr_wih_b,  49152);
  cvt_bf16_k<<<192, 256, 0, stream>>>(sr_whh,  sr_whh_b,  49152);
  cvt_bf16_k<<<384, 256, 0, stream>>>(dr_wih,  dr_wih_b,  98304);
  cvt_bf16_k<<<192, 256, 0, stream>>>(dr_whh,  dr_whh_b,  49152);
  cvt_bf16_k<<<4096, 256, 0, stream>>>(refs,   refs_b,    1048576);
  stmtcat_k<<<4096, 256, 0, stream>>>(statement, answer, stmtcat_b);

  // 1) context GRU encodes
  gru_wmma<<<1, 256, 0, stream>>>(stmtcat_b, ctx_wih_b, ctx_whh_b, ctx_bih, ctx_bhh,
                                  stmt_enc_b, nullptr, 256, 256);
  gru_wmma<<<8, 256, 0, stream>>>(refs_b, ctx_wih_b, ctx_whh_b, ctx_bih, ctx_bhh,
                                  docs_enc_b, nullptr, 128, 256);

  // 2) transposed operand copies (bf16)
  transpose_k<unsigned short><<<dim3(16, 16, 16), 256, 0, stream>>>(stmt_enc_b, stmt_encT_b, 256, 256);
  transpose_k<unsigned short><<<dim3(16, 8, 128), 256, 0, stream>>>(docs_enc_b, docs_encT_b, 128, 256);

  // 3) co-attention
  // match = stmt @ docs^T  [bon: 256x128, K=256] -> f32 (softmax input)
  gemm_bf16_wmma<<<dim3(4, 128), 256, 0, stream>>>(stmt_enc_b, docs_enc_b, match,
      128, 256, 65536LL, 32768LL, 32768LL, 8, 1, 32, 0);
  transpose_k<float><<<dim3(8, 16, 128), 256, 0, stream>>>(match, matchT, 256, 128);
  softmax_bf16_k<128><<<4096, 256, 0, stream>>>(match,  P1_b);   // over l
  softmax_bf16_k<256><<<2048, 256, 0, stream>>>(matchT, P2t_b);  // over s
  // read_sum = P1 @ docs  [bon: 256x256, K=128] -> bf16 (sr-GRU input)
  gemm_bf16_wmma<<<dim3(8, 128), 256, 0, stream>>>(P1_b, docs_encT_b, read_sum_b,
      256, 128, 32768LL, 32768LL, 65536LL, 1, 1, 64, 1);
  // doc_read = P2^T @ stmt [bon: 128x256, K=256] -> bf16 (dri half)
  gemm_bf16_wmma<<<dim3(4, 128), 256, 0, stream>>>(P2t_b, stmt_encT_b, doc_read_b,
      256, 256, 32768LL, 65536LL, 32768LL, 1, 8, 32, 1);

  // 4) doc-doc cross attention
  concat_dri_k<<<32768, 256, 0, stream>>>(docs_enc_b, doc_read_b, dri_b);
  transpose_k<unsigned short><<<dim3(32, 64, 16), 256, 0, stream>>>(dri_b, driT_b, 1024, 512);
  gemm_bf16_wmma<<<dim3(128, 16), 256, 0, stream>>>(dri_b, dri_b, mm,
      1024, 512, 524288LL, 524288LL, 1048576LL, 1, 1, 1024, 0);
  softmax_bf16_k<1024><<<2048, 256, 0, stream>>>(mm, P3_b);      // over m
  gemm_bf16_wmma<<<dim3(64, 16), 256, 0, stream>>>(P3_b, driT_b, att_b,
      512, 1024, 1048576LL, 524288LL, 524288LL, 1, 1, 512, 1);

  // 5) reasoning GRUs with fused max-pool over time
  gru_wmma<<<8, 256, 0, stream>>>(read_sum_b, sr_wih_b, sr_whh_b, sr_bih, sr_bhh,
                                  nullptr, s_r, 256, 256);
  gru_wmma<<<8, 256, 0, stream>>>(att_b, dr_wih_b, dr_whh_b, dr_bih, dr_bhh,
                                  nullptr, d_r, 128, 512);

  // 6) gate, pool over refs, score, softmax over options
  finalize_k<<<16, 256, 0, stream>>>(s_r, d_r, gate_w, gate_b, out_w, out_b, logits);
  softmax_out_k<<<1, 64, 0, stream>>>(logits, (float*)d_out);
}
